// sentenceEncoder_76501957476685
// MI455X (gfx1250) — matrix operand order repaired and checked by reading, not compiled, not run on hardware
//
#include <hip/hip_runtime.h>

// ---------------------------------------------------------------------------
// Bidirectional GRU sentence encoder for MI455X (gfx1250, wave32, WMMA).
// B=64, T=512, D_IN=300 (padded 320), H=512, 3H=1536, R=256, O=1.
// All GEMMs use v_wmma_f32_16x16x32_bf16 (f32 accumulate).
// Activation staging uses global_load_async_to_lds_b128 (ASYNCcnt path).
// ---------------------------------------------------------------------------

#define BATCH 64
#define TLEN  512
#define DIN   300
#define DPAD  320
#define HDIM  512
#define G3    1536
#define RDIM  256

typedef __attribute__((ext_vector_type(16))) __bf16 bf16x16;
typedef __attribute__((ext_vector_type(8)))  float  f32x8;

union Frag   { uint4 q[2]; bf16x16 v; };
union Half8  { uint4 q; unsigned short h[8]; };

__device__ __forceinline__ unsigned short f2bf(float f) {
    unsigned u = __float_as_uint(f);
    unsigned r = ((u >> 16) & 1u) + 0x7FFFu;      // round-to-nearest-even
    return (unsigned short)((u + r) >> 16);
}
__device__ __forceinline__ float bf2f(unsigned short h) {
    return __uint_as_float(((unsigned)h) << 16);
}

// Per-lane 16B async global->LDS copy (tracked by ASYNCcnt).
__device__ __forceinline__ void async_copy16(unsigned lds_off,
                                             const unsigned short* gptr) {
    asm volatile("global_load_async_to_lds_b128 %0, %1, off"
                 :: "v"(lds_off), "v"(gptr) : "memory");
}
__device__ __forceinline__ void async_wait0() {
    asm volatile("s_wait_asynccnt 0" ::: "memory");
}

// A-fragment (16x32 bf16, M x K) from row-major bf16 with row pitch `pitch`
// lanes 0-15: row=lane, K {kb..kb+7, kb+16..kb+23}
// lanes16-31: row=lane-16, K {kb+8..kb+15, kb+24..kb+31}
__device__ __forceinline__ bf16x16 load_a(const unsigned short* base, int pitch,
                                          int lane, int kb) {
    int r  = lane & 15;
    int k0 = kb + ((lane >> 4) << 3);
    const unsigned short* p = base + (size_t)r * pitch + k0;
    Frag f;
    f.q[0] = *(const uint4*)(p);
    f.q[1] = *(const uint4*)(p + 16);
    return f.v;
}

// B-fragment (32x16 bf16, K x N) from row-major [N][K] weights, row pitch `pitch`
// lane: N = lane&15, K {kb+(lane>>4)*8 + 0..7, kb+16+(lane>>4)*8 + 0..7}
__device__ __forceinline__ bf16x16 load_b(const unsigned short* wbase, int pitch,
                                          int lane, int kb) {
    int n  = lane & 15;
    int k0 = kb + ((lane >> 4) << 3);
    const unsigned short* p = wbase + (size_t)n * pitch + k0;
    Frag f;
    f.q[0] = *(const uint4*)(p);
    f.q[1] = *(const uint4*)(p + 16);
    return f.v;
}

// ---------------------------------------------------------------------------
// Kernel 1: fp32 -> bf16 convert with optional column padding
// ---------------------------------------------------------------------------
__global__ void convert_pad(const float* __restrict__ src,
                            unsigned short* __restrict__ dst,
                            int rows, int cin, int cout) {
    long i = (long)blockIdx.x * blockDim.x + threadIdx.x;
    long total = (long)rows * cout;
    if (i >= total) return;
    int r = (int)(i / cout), c = (int)(i % cout);
    dst[i] = (c < cin) ? f2bf(src[(long)r * cin + c]) : (unsigned short)0;
}

// ---------------------------------------------------------------------------
// Kernel 2: X [B,T,DIN] fp32 -> Xt [T,B,DPAD] bf16 (zero padded K)
// ---------------------------------------------------------------------------
__global__ void transpose_x(const float* __restrict__ X,
                            unsigned short* __restrict__ xt) {
    int t = blockIdx.x, b = blockIdx.y;
    for (int k = threadIdx.x; k < DPAD; k += blockDim.x) {
        unsigned short v = (k < DIN) ? f2bf(X[((size_t)b * TLEN + t) * DIN + k])
                                     : (unsigned short)0;
        xt[((size_t)t * BATCH + b) * DPAD + k] = v;
    }
}

// ---------------------------------------------------------------------------
// Kernel 3: input projection xi = X @ Wih^T + bih, both dirs.
// Grid (T, 2), block 1024 (32 waves). X slab staged in LDS via async copies.
// Output stored in C-fragment order: xi_frag[dir][t][mt(4)][nt(96)][lane][8]
// ---------------------------------------------------------------------------
__global__ __launch_bounds__(1024)
void input_proj(const unsigned short* __restrict__ xt,     // [T][64][320]
                const unsigned short* __restrict__ wih,    // [2][1536][320]
                const float* __restrict__ bih_f,
                const float* __restrict__ bih_b,
                unsigned short* __restrict__ xi_frag) {
    __shared__ unsigned short xtile[BATCH][336];   // pitch 336 halves (16B-aligned)

    int t = blockIdx.x, dir = blockIdx.y;
    int tid = threadIdx.x, lane = tid & 31, w = tid >> 5;
    const unsigned short* xrow0 = xt + (size_t)t * BATCH * DPAD;

    // Stage Xt[t] (64 x 320 bf16) into LDS with async 16B copies.
    unsigned lds0 = (unsigned)(unsigned long long)(void*)&xtile[0][0];
    for (int c = tid; c < BATCH * 40; c += 1024) {     // 40 x 16B per row
        int row = c / 40, cc = c % 40;
        async_copy16(lds0 + (unsigned)(row * 336 + cc * 8) * 2,
                     xrow0 + (size_t)row * DPAD + cc * 8);
    }
    async_wait0();
    __syncthreads();

    const unsigned short* wbase = wih + (size_t)dir * G3 * DPAD;
    const float* bih = dir ? bih_b : bih_f;
    int r = lane & 15;
    int koffA = (lane >> 4) << 3;

#pragma unroll 1
    for (int j = 0; j < 3; ++j) {
        int nt = w * 3 + j;
        const unsigned short* wrow = wbase + (size_t)(nt * 16 + r) * DPAD;
        float bias = bih[nt * 16 + r];
#pragma unroll 1
        for (int mt = 0; mt < 4; ++mt) {
            int rowi = mt * 16 + r;
            f32x8 c = {};
#pragma unroll 2
            for (int k = 0; k < 10; ++k) {
                Frag fa;
                fa.q[0] = *(const uint4*)(&xtile[rowi][k * 32 + koffA]);
                fa.q[1] = *(const uint4*)(&xtile[rowi][k * 32 + koffA + 16]);
                bf16x16 fb = load_b(wrow, DPAD, lane, k * 32);
                c = __builtin_amdgcn_wmma_f32_16x16x32_bf16(
                        false, fa.v, false, fb, (short)0, c, false, false);
            }
            Half8 o;
#pragma unroll
            for (int v = 0; v < 8; ++v) o.h[v] = f2bf(c[v] + bias);
            *(uint4*)(xi_frag +
                ((((size_t)dir * TLEN + t) * 4 + mt) * 96 + nt) * 256 + lane * 8) = o.q;
        }
    }
}

// ---------------------------------------------------------------------------
// One gate's GEMM slice: h(LDS, 16x512 bf16) @ Whh^T rows -> 16x16 f32 tile.
// No operand arrays -> nothing spillable; unroll capped at 4.
// ---------------------------------------------------------------------------
__device__ __forceinline__ f32x8 gate_gemm(const unsigned short (*h_bf)[528],
                                           const unsigned short* wrow,
                                           int lane) {
    int r = lane & 15;
    int koffA = (lane >> 4) << 3;
    int koffB = (lane >> 4) << 3;
    f32x8 c = {};
#pragma unroll 4
    for (int k = 0; k < HDIM; k += 32) {
        Frag fa, fb;
        fa.q[0] = *(const uint4*)(&h_bf[r][k + koffA]);
        fa.q[1] = *(const uint4*)(&h_bf[r][k + koffA + 16]);
        const unsigned short* p = wrow + k + koffB;
        fb.q[0] = *(const uint4*)(p);
        fb.q[1] = *(const uint4*)(p + 16);
        c = __builtin_amdgcn_wmma_f32_16x16x32_bf16(
                false, fa.v, false, fb.v, (short)0, c, false, false);
    }
    return c;
}

// ---------------------------------------------------------------------------
// Kernel 4: GRU recurrence. Grid (4 batch-tiles, 2 dirs), block 1024 (32 waves).
// Persistent workgroup loops over 512 time steps; h resident in LDS
// (bf16 copy for WMMA-A, f32 copy for gate math). Whh streamed from L2.
// Wave jt owns hidden columns [jt*16, jt*16+16), computes gates r/z/n.
// ---------------------------------------------------------------------------
__global__ __launch_bounds__(1024)
void gru_recurrence(const unsigned short* __restrict__ whh_bf,  // [2][1536][512]
                    const float* __restrict__ bhh_f,
                    const float* __restrict__ bhh_b,
                    const unsigned short* __restrict__ xi_frag,
                    const int* __restrict__ lens,
                    unsigned short* __restrict__ hcat) {        // [64][512][1024]
    __shared__ unsigned short h_bf[16][528];   // pitch 528 halves (16B-aligned)
    __shared__ float          h_f32[16][516];
    __shared__ float          bhh_s[G3];
    __shared__ int            len_s[16];

    int mt = blockIdx.x, dir = blockIdx.y;
    int tid = threadIdx.x, lane = tid & 31, jt = tid >> 5;

    for (int i = tid; i < 16 * 512; i += 1024) {
        int r = i >> 9, c = i & 511;
        h_bf[r][c] = 0;
        h_f32[r][c] = 0.f;
    }
    const float* bhh = dir ? bhh_b : bhh_f;
    for (int i = tid; i < G3; i += 1024) bhh_s[i] = bhh[i];
    if (tid < 16) len_s[tid] = lens[mt * 16 + tid];
    __syncthreads();

    const unsigned short* whh = whh_bf + (size_t)dir * G3 * HDIM;
    int col = jt * 16 + (lane & 15);
    int mr0 = (lane < 16) ? 0 : 8;
    const unsigned short* wr_r = whh + (size_t)((0 * 32 + jt) * 16 + (lane & 15)) * HDIM;
    const unsigned short* wr_z = whh + (size_t)((1 * 32 + jt) * 16 + (lane & 15)) * HDIM;
    const unsigned short* wr_n = whh + (size_t)((2 * 32 + jt) * 16 + (lane & 15)) * HDIM;
    float bhr = bhh_s[col], bhz = bhh_s[HDIM + col], bhn = bhh_s[2 * HDIM + col];

    for (int tt = 0; tt < TLEN; ++tt) {
        int ta = dir ? (TLEN - 1 - tt) : tt;

        f32x8 ar = gate_gemm(h_bf, wr_r, lane);
        f32x8 az = gate_gemm(h_bf, wr_z, lane);
        f32x8 an = gate_gemm(h_bf, wr_n, lane);

        // xi fragments (C-layout match) for the three gates
        const unsigned short* xib =
            xi_frag + ((((size_t)dir * TLEN + ta) * 4 + mt) * 96) * 256;
        Half8 xr, xz, xn;
        xr.q = *(const uint4*)(xib + (size_t)(0 * 32 + jt) * 256 + lane * 8);
        xz.q = *(const uint4*)(xib + (size_t)(1 * 32 + jt) * 256 + lane * 8);
        xn.q = *(const uint4*)(xib + (size_t)(2 * 32 + jt) * 256 + lane * 8);

        float hn[8];
#pragma unroll
        for (int v = 0; v < 8; ++v) {
            int m = v + mr0;
            float gr = ar[v] + bhr + bf2f(xr.h[v]);
            float gz = az[v] + bhz + bf2f(xz.h[v]);
            float rg = 1.f / (1.f + __expf(-gr));
            float zg = 1.f / (1.f + __expf(-gz));
            float ng = tanhf(bf2f(xn.h[v]) + rg * (an[v] + bhn));
            float hp = h_f32[m][col];
            float hc = (1.f - zg) * ng + zg * hp;
            bool valid = ta < len_s[m];
            float hnew = valid ? hc : hp;
            hn[v] = hnew;
            hcat[(((size_t)(mt * 16 + m)) * TLEN + ta) * (2 * HDIM)
                 + dir * HDIM + col] = valid ? f2bf(hnew) : (unsigned short)0;
        }
        __syncthreads();           // all waves done reading old h
#pragma unroll
        for (int v = 0; v < 8; ++v) {
            int m = v + mr0;
            h_f32[m][col] = hn[v];
            h_bf[m][col]  = f2bf(hn[v]);
        }
        __syncthreads();           // new h visible before next step
    }
}

// ---------------------------------------------------------------------------
// Kernel 5: attention pooling + output head. One block (256 thr, 8 waves)
// per batch element. Y = tanh(Hcat@g2r_W^T + b) via WMMA, scores = Y@u_s,
// masked softmax over T, pooled = w^T Hcat, out = sigmoid(pooled@r2o_W^T+b).
// ---------------------------------------------------------------------------
__global__ __launch_bounds__(256)
void attention(const unsigned short* __restrict__ hcat,   // [64][512][1024]
               const unsigned short* __restrict__ g2rw,   // [256][1024] bf16
               const float* __restrict__ g2rb,
               const float* __restrict__ u_s,
               const float* __restrict__ r2ow,            // [1][1024]
               const float* __restrict__ r2ob,
               const int* __restrict__ lens,
               float* __restrict__ out) {
    __shared__ float sc[TLEN];
    __shared__ float wgt[TLEN];
    __shared__ float red[256];

    int b = blockIdx.x, tid = threadIdx.x, lane = tid & 31, w = tid >> 5;
    for (int i = tid; i < TLEN; i += 256) sc[i] = 0.f;
    __syncthreads();

    const unsigned short* hb = hcat + (size_t)b * TLEN * (2 * HDIM);
    int colr = lane & 15;
    int mr0  = (lane < 16) ? 0 : 8;

    // scores via WMMA: each wave handles 4 row-tiles of 16 time steps
#pragma unroll 1
    for (int mti = 0; mti < 4; ++mti) {
        int mtile = w * 4 + mti;
        const unsigned short* abase = hb + (size_t)mtile * 16 * (2 * HDIM);
        float rowsum[8];
#pragma unroll
        for (int v = 0; v < 8; ++v) rowsum[v] = 0.f;
#pragma unroll 1
        for (int nt = 0; nt < 16; ++nt) {
            const unsigned short* wrow = g2rw + (size_t)(nt * 16 + colr) * (2 * HDIM);
            f32x8 c = {};
#pragma unroll 4
            for (int k = 0; k < 2 * HDIM; k += 32) {
                bf16x16 fa = load_a(abase, 2 * HDIM, lane, k);
                bf16x16 fb = load_b(wrow, 2 * HDIM, lane, k);
                c = __builtin_amdgcn_wmma_f32_16x16x32_bf16(
                        false, fa, false, fb, (short)0, c, false, false);
            }
            float bias = g2rb[nt * 16 + colr];
            float us   = u_s[nt * 16 + colr];
#pragma unroll
            for (int v = 0; v < 8; ++v)
                rowsum[v] += tanhf(c[v] + bias) * us;
        }
#pragma unroll
        for (int v = 0; v < 8; ++v)
            atomicAdd(&sc[mtile * 16 + mr0 + v], rowsum[v]);
    }
    __syncthreads();

    // masked softmax over T
    int len = lens[b];
    for (int t = tid; t < TLEN; t += 256)
        if (t >= len) sc[t] = -1e30f;
    __syncthreads();

    float lm = -1e30f;
    for (int t = tid; t < TLEN; t += 256) lm = fmaxf(lm, sc[t]);
    red[tid] = lm; __syncthreads();
    for (int s = 128; s > 0; s >>= 1) {
        if (tid < s) red[tid] = fmaxf(red[tid], red[tid + s]);
        __syncthreads();
    }
    float mx = red[0];
    __syncthreads();

    float ls = 0.f;
    for (int t = tid; t < TLEN; t += 256) {
        float e = __expf(sc[t] - mx);
        wgt[t] = e;
        ls += e;
    }
    red[tid] = ls; __syncthreads();
    for (int s = 128; s > 0; s >>= 1) {
        if (tid < s) red[tid] += red[tid + s];
        __syncthreads();
    }
    float inv = 1.f / red[0];
    __syncthreads();

    // pooled = sum_t w_t * Hcat[b,t,:]; thread owns 4 hidden columns
    float p0 = 0.f, p1 = 0.f, p2 = 0.f, p3 = 0.f;
    for (int t = 0; t < TLEN; ++t) {
        float wt = wgt[t] * inv;
        const unsigned short* row = hb + (size_t)t * (2 * HDIM);
        p0 += wt * bf2f(row[tid]);
        p1 += wt * bf2f(row[tid + 256]);
        p2 += wt * bf2f(row[tid + 512]);
        p3 += wt * bf2f(row[tid + 768]);
    }
    float o = p0 * r2ow[tid] + p1 * r2ow[tid + 256] +
              p2 * r2ow[tid + 512] + p3 * r2ow[tid + 768];
    red[tid] = o; __syncthreads();
    for (int s = 128; s > 0; s >>= 1) {
        if (tid < s) red[tid] += red[tid + s];
        __syncthreads();
    }
    if (tid == 0)
        out[b] = 1.f / (1.f + __expf(-(red[0] + r2ob[0])));
}

// ---------------------------------------------------------------------------
// Host launcher
// ---------------------------------------------------------------------------
extern "C" void kernel_launch(void* const* d_in, const int* in_sizes, int n_in,
                              void* d_out, int out_size, void* d_ws, size_t ws_size,
                              hipStream_t stream) {
    const float* X     = (const float*)d_in[0];
    const int*   lens  = (const int*)d_in[1];
    const float* Wih_f = (const float*)d_in[3];
    const float* Whh_f = (const float*)d_in[4];
    const float* bih_f = (const float*)d_in[5];
    const float* bhh_f = (const float*)d_in[6];
    const float* Wih_b = (const float*)d_in[7];
    const float* Whh_b = (const float*)d_in[8];
    const float* bih_b = (const float*)d_in[9];
    const float* bhh_b = (const float*)d_in[10];
    const float* g2r_W = (const float*)d_in[11];
    const float* g2r_b = (const float*)d_in[12];
    const float* u_s   = (const float*)d_in[13];
    const float* r2o_W = (const float*)d_in[14];
    const float* r2o_b = (const float*)d_in[15];

    char* ws = (char*)d_ws;
    size_t off = 0;
    auto take = [&](size_t bytes) -> void* {
        void* p = ws + off;
        off = (off + bytes + 255) & ~(size_t)255;
        return p;
    };
    unsigned short* xt     = (unsigned short*)take((size_t)TLEN * BATCH * DPAD * 2);
    unsigned short* wih    = (unsigned short*)take((size_t)2 * G3 * DPAD * 2);
    unsigned short* whh    = (unsigned short*)take((size_t)2 * G3 * HDIM * 2);
    unsigned short* g2rw   = (unsigned short*)take((size_t)RDIM * 2 * HDIM * 2);
    unsigned short* xifrag = (unsigned short*)take((size_t)2 * TLEN * 4 * 96 * 256 * 2);
    unsigned short* hcat   = (unsigned short*)take((size_t)BATCH * TLEN * 2 * HDIM * 2);

    auto blocks = [](long n) { return (unsigned)((n + 255) / 256); };

    convert_pad<<<blocks((long)G3 * DPAD), 256, 0, stream>>>(Wih_f, wih, G3, DIN, DPAD);
    convert_pad<<<blocks((long)G3 * DPAD), 256, 0, stream>>>(Wih_b, wih + (size_t)G3 * DPAD, G3, DIN, DPAD);
    convert_pad<<<blocks((long)G3 * HDIM), 256, 0, stream>>>(Whh_f, whh, G3, HDIM, HDIM);
    convert_pad<<<blocks((long)G3 * HDIM), 256, 0, stream>>>(Whh_b, whh + (size_t)G3 * HDIM, G3, HDIM, HDIM);
    convert_pad<<<blocks((long)RDIM * 2 * HDIM), 256, 0, stream>>>(g2r_W, g2rw, RDIM, 2 * HDIM, 2 * HDIM);

    transpose_x<<<dim3(TLEN, BATCH), 256, 0, stream>>>(X, xt);

    input_proj<<<dim3(TLEN, 2), 1024, 0, stream>>>(xt, wih, bih_f, bih_b, xifrag);

    gru_recurrence<<<dim3(4, 2), 1024, 0, stream>>>(whh, bhh_f, bhh_b, xifrag, lens, hcat);

    attention<<<64, 256, 0, stream>>>(hcat, g2rw, g2r_b, u_s, r2o_W, r2o_b, lens,
                                      (float*)d_out);
}
